// CoDeX_Net_37761352467128
// MI455X (gfx1250) — compile-verified, matrix operand order checked by reading, not run on hardware
//
#include <hip/hip_runtime.h>
#include <hip/hip_bf16.h>
#include <math.h>

#define B_   8
#define C_   32
#define H_   256
#define W_   256
#define K_   4
#define HW_  (H_ * W_)
#define RMAX 2.0f

typedef _Float16 v8h  __attribute__((ext_vector_type(8)));
typedef _Float16 v16h __attribute__((ext_vector_type(16)));
typedef float    v8f  __attribute__((ext_vector_type(8)));

static __device__ inline v8f wmma16(v16h a, v16h b, v8f c) {
  return __builtin_amdgcn_wmma_f32_16x16x32_f16(false, a, false, b, (short)0, c,
                                                false, false);
}

// Fragment loader per CDNA5 16-bit 16x32 layout:
// lane l (hi = l>=16) holds K in [hi*8, hi*8+8) and [hi*8+16, hi*8+24).
// Works for LDS or global rows of 32 halfs (16B aligned chunks -> b128 loads).
static __device__ inline v16h frag_from_h(const _Float16* row, int hi) {
  v8h lo = *(const v8h*)(row + hi * 8);
  v8h hu = *(const v8h*)(row + hi * 8 + 16);
  return __builtin_shufflevector(lo, hu, 0, 1, 2, 3, 4, 5, 6, 7, 8, 9, 10, 11,
                                 12, 13, 14, 15);
}

static __device__ inline v16h frag_from_f32(const float* row, int hi) {
  v16h a;
  const float* p0 = row + hi * 8;
  const float* p1 = row + hi * 8 + 16;
#pragma unroll
  for (int e = 0; e < 8; ++e) {
    a[e]     = (_Float16)p0[e];
    a[8 + e] = (_Float16)p1[e];
  }
  return a;
}

// conv 3x3 weight fragment: element i lives at base[i*9] (OIHW, tap folded in)
static __device__ inline v16h frag_from_c1(const float* base, int hi) {
  v16h b;
#pragma unroll
  for (int e = 0; e < 8; ++e) {
    b[e]     = (_Float16)base[(hi * 8 + e) * 9];
    b[8 + e] = (_Float16)base[(hi * 8 + 16 + e) * 9];
  }
  return b;
}

static __device__ inline float silu_(float y) { return y / (1.f + __expf(-y)); }

// ---------------------------------------------------------------- pooling ---
__global__ void pool2_kernel(const float* __restrict__ in, float* __restrict__ out,
                             int Hin, int Win) {
  int Ho = Hin >> 1, Wo = Win >> 1;
  int idx = blockIdx.x * blockDim.x + threadIdx.x;
  int total = B_ * C_ * Ho * Wo;
  if (idx >= total) return;
  int wo = idx % Wo;
  int t = idx / Wo;
  int ho = t % Ho;
  t /= Ho;
  int c = t % C_;
  int b = t / C_;
  const float* p = in + (((size_t)b * C_ + c) * Hin + 2 * ho) * Win + 2 * wo;
  out[idx] = 0.25f * (p[0] + p[1] + p[Win] + p[Win + 1]);
}

// -------------------------------------------- depthwise-separable + BN+SiLU -
// input NCHW fp32, output NHWC f16.  blockDim = 2*CHUNK, one row-chunk/block.
template <int CHUNK>
__global__ void ds_kernel(const float* __restrict__ xin, const float* __restrict__ dw,
                          const float* __restrict__ pw, const float* __restrict__ gam,
                          const float* __restrict__ bet, const float* __restrict__ mu,
                          const float* __restrict__ var, _Float16* __restrict__ zout,
                          int Hin, int Win) {
  __shared__ __align__(16) float xs[C_][3][CHUNK + 2];
  __shared__ __align__(16) _Float16 ts[CHUNK][C_];
  const int tid = threadIdx.x;
  const int b = blockIdx.z, h = blockIdx.y, w0 = blockIdx.x * CHUNK;

  for (int i = tid; i < C_ * 3 * (CHUNK + 2); i += blockDim.x) {
    int col = i % (CHUNK + 2);
    int t = i / (CHUNK + 2);
    int r = t % 3;
    int c = t / 3;
    int hh = h + r - 1, ww = w0 + col - 1;
    float v = 0.f;
    if (hh >= 0 && hh < Hin && ww >= 0 && ww < Win)
      v = xin[(((size_t)b * C_ + c) * Hin + hh) * Win + ww];
    xs[c][r][col] = v;
  }
  __syncthreads();

  for (int i = tid; i < CHUNK * C_; i += blockDim.x) {
    int p = i % CHUNK, c = i / CHUNK;
    const float* wv = dw + c * 9;
    float acc = 0.f;
#pragma unroll
    for (int r = 0; r < 3; ++r)
#pragma unroll
      for (int s = 0; s < 3; ++s) acc += xs[c][r][p + s] * wv[r * 3 + s];
    ts[p][c] = (_Float16)acc;
  }
  __syncthreads();

  const int lane = tid & 31, wave = tid >> 5;
  const int hi = lane >> 4, n = lane & 15;
  v16h A = frag_from_h(&ts[wave * 16 + n][0], hi);
  float yv[2][8];
#pragma unroll
  for (int nt = 0; nt < 2; ++nt) {
    int o = n + nt * 16;
    v16h Bw = frag_from_f32(pw + o * C_, hi);
    v8f acc = {0, 0, 0, 0, 0, 0, 0, 0};
    acc = wmma16(A, Bw, acc);
    float sc = gam[o] * rsqrtf(var[o] + 1e-5f);
    float sh = bet[o] - mu[o] * sc;
#pragma unroll
    for (int r = 0; r < 8; ++r) yv[nt][r] = silu_(acc[r] * sc + sh);
  }
  __syncthreads();  // all waves done reading ts as A
#pragma unroll
  for (int nt = 0; nt < 2; ++nt)
#pragma unroll
    for (int r = 0; r < 8; ++r)
      ts[wave * 16 + hi * 8 + r][n + nt * 16] = (_Float16)yv[nt][r];
  __syncthreads();
  // coalesced NHWC f16 store: 8 halfs (16B) per unit
  const size_t pixBase = ((size_t)b * Hin + h) * Win + w0;
  for (int u = tid; u < CHUNK * 4; u += blockDim.x) {
    int p = u >> 2, c8 = u & 3;
    *(v8h*)(zout + (pixBase + p) * C_ + c8 * 8) = *(const v8h*)(&ts[p][c8 * 8]);
  }
}

// ----------------------------------------------------- pointwise (NHWC) -----
// f16 NHWC in -> f16 NHWC out, via LDS transpose for coalesced stores.
__global__ void pw_nhwc_kernel(const _Float16* __restrict__ zin,
                               const float* __restrict__ wmat,
                               _Float16* __restrict__ qout, int HW) {
  __shared__ __align__(16) _Float16 ts[128][C_];
  const int tid = threadIdx.x, lane = tid & 31, wave = tid >> 5;
  const int hi = lane >> 4, n = lane & 15;
  const int b = blockIdx.z;
  const size_t blk0 = (size_t)blockIdx.x * 128;
  const size_t pix0 = blk0 + wave * 16;
  v16h A = frag_from_h(zin + ((size_t)b * HW + pix0 + n) * C_, hi);
#pragma unroll
  for (int nt = 0; nt < 2; ++nt) {
    int o = n + nt * 16;
    v16h Bw = frag_from_f32(wmat + o * C_, hi);
    v8f acc = {0, 0, 0, 0, 0, 0, 0, 0};
    acc = wmma16(A, Bw, acc);
#pragma unroll
    for (int r = 0; r < 8; ++r)
      ts[wave * 16 + hi * 8 + r][o] = (_Float16)acc[r];
  }
  __syncthreads();
  const size_t rowBase = (size_t)b * HW + blk0;
  for (int u = tid; u < 128 * 4; u += 256) {
    int p = u >> 2, c8 = u & 3;
    *(v8h*)(qout + (rowBase + p) * C_ + c8 * 8) = *(const v8h*)(&ts[p][c8 * 8]);
  }
}

// --------------------------------------------------------------- head ------
// conv3x3(32->16)+SiLU via 9 accumulating WMMAs, then 1x1(16->12),
// tanh*RMAX offsets + softmax weights.  q is f16 NHWC.  Output: (B,HW,12) f32
__global__ void head_kernel(const _Float16* __restrict__ q,
                            const float* __restrict__ c1,
                            const float* __restrict__ c2w,
                            const float* __restrict__ c2b, float* __restrict__ hout) {
  __shared__ __align__(16) _Float16 qs[3][130][C_];
  __shared__ __align__(16) _Float16 hs[128][16];
  const int tid = threadIdx.x;
  const int b = blockIdx.z, h = blockIdx.y, w0 = blockIdx.x * 128;

  for (int u = tid; u < 3 * 130 * 4; u += 256) {
    int c8 = u & 3;
    int t = u >> 2;
    int col = t % 130;
    int r = t / 130;
    int hh = h + r - 1, ww = w0 + col - 1;
    v8h v = {0, 0, 0, 0, 0, 0, 0, 0};
    if (hh >= 0 && hh < H_ && ww >= 0 && ww < W_)
      v = *(const v8h*)(q + ((((size_t)b * H_ + hh) * W_ + ww) * C_) + c8 * 8);
    *(v8h*)(&qs[r][col][c8 * 8]) = v;
  }
  __syncthreads();

  const int lane = tid & 31, wave = tid >> 5, hi = lane >> 4, n = lane & 15;
  v8f acc = {0, 0, 0, 0, 0, 0, 0, 0};
#pragma unroll
  for (int ty = 0; ty < 3; ++ty)
#pragma unroll
    for (int tx = 0; tx < 3; ++tx) {
      v16h A = frag_from_h(&qs[ty][wave * 16 + n + tx][0], hi);
      v16h Bw = frag_from_c1(c1 + n * (C_ * 9) + (ty * 3 + tx), hi);
      acc = wmma16(A, Bw, acc);
    }
#pragma unroll
  for (int r = 0; r < 8; ++r)
    hs[wave * 16 + hi * 8 + r][n] = (_Float16)silu_(acc[r]);
  __syncthreads();

  if (tid < 128) {
    float o[12];
#pragma unroll
    for (int j = 0; j < 12; ++j) {
      float s = c2b[j];
#pragma unroll
      for (int i = 0; i < 16; ++i) s += (float)hs[tid][i] * c2w[j * 16 + i];
      o[j] = s;
    }
    float res[12];
#pragma unroll
    for (int k = 0; k < K_; ++k) {
      res[2 * k] = RMAX * tanhf(o[2 * k]);
      res[2 * k + 1] = RMAX * tanhf(o[2 * k + 1]);
    }
    float mx = fmaxf(fmaxf(o[8], o[9]), fmaxf(o[10], o[11]));
    float e0 = __expf(o[8] - mx), e1 = __expf(o[9] - mx);
    float e2 = __expf(o[10] - mx), e3 = __expf(o[11] - mx);
    float inv = 1.f / (e0 + e1 + e2 + e3);
    res[8] = e0 * inv; res[9] = e1 * inv; res[10] = e2 * inv; res[11] = e3 * inv;
    size_t p = ((size_t)b * H_ + h) * W_ + w0 + tid;
#pragma unroll
    for (int j = 0; j < 12; ++j) hout[p * 12 + j] = res[j];
  }
}

// --------------------------------------------------------------- zero ------
__global__ void zero_kernel(float* p, int n) {
  int i = blockIdx.x * blockDim.x + threadIdx.x;
  if (i < n) p[i] = 0.f;
}

// --------------------------------------- mean over pixels of f16 NHWC -------
__global__ void mean_nhwc_kernel(const _Float16* __restrict__ z,
                                 float* __restrict__ means, int moff) {
  __shared__ float red[256][C_];
  const int tid = threadIdx.x, b = blockIdx.z;
  float acc[C_];
#pragma unroll
  for (int c = 0; c < C_; ++c) acc[c] = 0.f;
  for (int it = 0; it < 16; ++it) {
    size_t p = (size_t)b * HW_ + (size_t)blockIdx.x * 4096 + it * 256 + tid;
    const v8h* r = (const v8h*)(z + p * C_);
#pragma unroll
    for (int c8 = 0; c8 < 4; ++c8) {
      v8h v = r[c8];
#pragma unroll
      for (int e = 0; e < 8; ++e) acc[c8 * 8 + e] += (float)v[e];
    }
  }
#pragma unroll
  for (int c = 0; c < C_; ++c) red[tid][c] = acc[c];
  __syncthreads();
  for (int s = 128; s > 0; s >>= 1) {
    if (tid < s)
      for (int c = 0; c < C_; ++c) red[tid][c] += red[tid + s][c];
    __syncthreads();
  }
  if (tid < C_) atomicAdd(&means[b * 96 + moff + tid], red[0][tid]);
}

// ---------------------------------------------------- deformable sampling ---
__global__ void sample_kernel(const _Float16* __restrict__ z, int Hc, int Wc,
                              const float* __restrict__ hd,
                              _Float16* __restrict__ y,
                              float* __restrict__ means, int moff) {
  __shared__ float red[256][C_];
  const int tid = threadIdx.x, b = blockIdx.z;
  const int p = blockIdx.x * 256 + tid;
  const int wf = p & (W_ - 1), hf = p >> 8;
  const float xc = (wf + 0.5f) * ((float)Wc / (float)W_) - 0.5f;
  const float yc = (hf + 0.5f) * ((float)Hc / (float)H_) - 0.5f;
  const float* hp = hd + ((size_t)b * HW_ + p) * 12;
  const _Float16* zb = z + (size_t)b * Hc * Wc * C_;
  float acc[C_];
#pragma unroll
  for (int c = 0; c < C_; ++c) acc[c] = 0.f;
#pragma unroll
  for (int k = 0; k < K_; ++k) {
    float xs = fminf(fmaxf(xc + hp[2 * k], 0.f), (float)(Wc - 1));
    float ys = fminf(fmaxf(yc + hp[2 * k + 1], 0.f), (float)(Hc - 1));
    float x0f = floorf(xs), y0f = floorf(ys);
    float tx = xs - x0f, ty = ys - y0f;
    int x0 = (int)x0f, y0 = (int)y0f;
    int x1 = min(x0 + 1, Wc - 1), y1 = min(y0 + 1, Hc - 1);
    float wk = hp[8 + k];
    float w00 = wk * (1.f - tx) * (1.f - ty), w01 = wk * tx * (1.f - ty);
    float w10 = wk * (1.f - tx) * ty, w11 = wk * tx * ty;
    const v8h* p00 = (const v8h*)(zb + ((size_t)y0 * Wc + x0) * C_);
    const v8h* p01 = (const v8h*)(zb + ((size_t)y0 * Wc + x1) * C_);
    const v8h* p10 = (const v8h*)(zb + ((size_t)y1 * Wc + x0) * C_);
    const v8h* p11 = (const v8h*)(zb + ((size_t)y1 * Wc + x1) * C_);
#pragma unroll
    for (int c8 = 0; c8 < 4; ++c8) {
      v8h h00 = p00[c8], h01 = p01[c8], h10 = p10[c8], h11 = p11[c8];
#pragma unroll
      for (int e = 0; e < 8; ++e)
        acc[c8 * 8 + e] += w00 * (float)h00[e] + w01 * (float)h01[e] +
                           w10 * (float)h10[e] + w11 * (float)h11[e];
    }
  }
  v8h* yo = (v8h*)(y + ((size_t)b * HW_ + p) * C_);
#pragma unroll
  for (int c8 = 0; c8 < 4; ++c8) {
    v8h o;
#pragma unroll
    for (int e = 0; e < 8; ++e) o[e] = (_Float16)acc[c8 * 8 + e];
    yo[c8] = o;
  }
#pragma unroll
  for (int c = 0; c < C_; ++c) red[tid][c] = acc[c];
  __syncthreads();
  for (int s = 128; s > 0; s >>= 1) {
    if (tid < s)
      for (int c = 0; c < C_; ++c) red[tid][c] += red[tid + s][c];
    __syncthreads();
  }
  if (tid < C_) atomicAdd(&means[b * 96 + moff + tid], red[0][tid]);
}

// ------------------------------------------------------------- gate MLP -----
__global__ void alpha_kernel(const float* __restrict__ means,
                             const float* __restrict__ w1, const float* __restrict__ b1,
                             const float* __restrict__ w2, const float* __restrict__ b2,
                             float* __restrict__ alpha) {
  int b = threadIdx.x;
  if (b >= B_) return;
  const float inv = 1.f / (float)HW_;
  float d[96];
  for (int j = 0; j < 96; ++j) d[j] = means[b * 96 + j] * inv;
  float hb[32];
  for (int o = 0; o < 32; ++o) {
    float s = b1[o];
    for (int j = 0; j < 96; ++j) s += d[j] * w1[o * 96 + j];
    hb[o] = silu_(s);
  }
  float a[3];
  for (int i = 0; i < 3; ++i) {
    float s = b2[i];
    for (int o = 0; o < 32; ++o) s += hb[o] * w2[i * 32 + o];
    a[i] = s;
  }
  float mx = fmaxf(a[0], fmaxf(a[1], a[2]));
  float e0 = __expf(a[0] - mx), e1 = __expf(a[1] - mx), e2 = __expf(a[2] - mx);
  float is = 1.f / (e0 + e1 + e2);
  alpha[b * 3 + 0] = e0 * is;
  alpha[b * 3 + 1] = e1 * is;
  alpha[b * 3 + 2] = e2 * is;
}

// -------------------------------------- final mix + pointwise + residual ----
__global__ void final_kernel(const _Float16* __restrict__ z0,
                             const _Float16* __restrict__ y1,
                             const _Float16* __restrict__ y2,
                             const float* __restrict__ alpha,
                             const float* __restrict__ wmat, const float* __restrict__ x,
                             float* __restrict__ out) {
  const int tid = threadIdx.x, lane = tid & 31, wave = tid >> 5;
  const int hi = lane >> 4, n = lane & 15;
  const int b = blockIdx.z;
  const size_t pix0 = (size_t)blockIdx.x * 128 + wave * 16;
  const float a0 = alpha[b * 3 + 0], a1 = alpha[b * 3 + 1], a2 = alpha[b * 3 + 2];
  const size_t rowOff = ((size_t)b * HW_ + pix0 + n) * C_;
  const _Float16* p0 = z0 + rowOff;
  const _Float16* p1 = y1 + rowOff;
  const _Float16* p2 = y2 + rowOff;
  v8h z_lo = *(const v8h*)(p0 + hi * 8), z_hi = *(const v8h*)(p0 + hi * 8 + 16);
  v8h u_lo = *(const v8h*)(p1 + hi * 8), u_hi = *(const v8h*)(p1 + hi * 8 + 16);
  v8h v_lo = *(const v8h*)(p2 + hi * 8), v_hi = *(const v8h*)(p2 + hi * 8 + 16);
  v16h A;
#pragma unroll
  for (int e = 0; e < 8; ++e) {
    A[e]     = (_Float16)(a0 * (float)z_lo[e] + a1 * (float)u_lo[e] + a2 * (float)v_lo[e]);
    A[8 + e] = (_Float16)(a0 * (float)z_hi[e] + a1 * (float)u_hi[e] + a2 * (float)v_hi[e]);
  }
#pragma unroll
  for (int nt = 0; nt < 2; ++nt) {
    int o = n + nt * 16;
    v16h Bw = frag_from_f32(wmat + o * C_, hi);
    v8f acc = {0, 0, 0, 0, 0, 0, 0, 0};
    acc = wmma16(A, Bw, acc);
    size_t obase = ((size_t)(b * C_ + o)) * HW_ + pix0 + hi * 8;
#pragma unroll
    for (int r = 0; r < 8; ++r) out[obase + r] = x[obase + r] + acc[r];
  }
}

// ---------------------------------------------------------------------------
extern "C" void kernel_launch(void* const* d_in, const int* in_sizes, int n_in,
                              void* d_out, int out_size, void* d_ws, size_t ws_size,
                              hipStream_t stream) {
  (void)in_sizes; (void)n_in; (void)out_size; (void)ws_size;
  const float* x       = (const float*)d_in[0];
  const float* ds_dw[3] = {(const float*)d_in[1], (const float*)d_in[7],  (const float*)d_in[13]};
  const float* ds_pw[3] = {(const float*)d_in[2], (const float*)d_in[8],  (const float*)d_in[14]};
  const float* ds_g[3]  = {(const float*)d_in[3], (const float*)d_in[9],  (const float*)d_in[15]};
  const float* ds_b[3]  = {(const float*)d_in[4], (const float*)d_in[10], (const float*)d_in[16]};
  const float* ds_m[3]  = {(const float*)d_in[5], (const float*)d_in[11], (const float*)d_in[17]};
  const float* ds_v[3]  = {(const float*)d_in[6], (const float*)d_in[12], (const float*)d_in[18]};
  const float* qproj_w = (const float*)d_in[19];
  const float* h1_c1 = (const float*)d_in[20];
  const float* h1_c2w = (const float*)d_in[21];
  const float* h1_c2b = (const float*)d_in[22];
  const float* h2_c1 = (const float*)d_in[23];
  const float* h2_c2w = (const float*)d_in[24];
  const float* h2_c2b = (const float*)d_in[25];
  const float* r_w1 = (const float*)d_in[26];
  const float* r_b1 = (const float*)d_in[27];
  const float* r_w2 = (const float*)d_in[28];
  const float* r_b2 = (const float*)d_in[29];
  const float* final_w = (const float*)d_in[30];

  char* base = (char*)d_ws;
  size_t off = 0;
  auto alloc = [&](size_t bytes) -> char* {
    char* p = base + off;
    off += (bytes + 255) & ~(size_t)255;
    return p;
  };
  float*    s1  = (float*)alloc((size_t)B_ * C_ * 128 * 128 * 4);
  float*    s2  = (float*)alloc((size_t)B_ * C_ * 64 * 64 * 4);
  _Float16* z0  = (_Float16*)alloc((size_t)B_ * HW_ * C_ * 2);
  _Float16* z1  = (_Float16*)alloc((size_t)B_ * 128 * 128 * C_ * 2);
  _Float16* z2  = (_Float16*)alloc((size_t)B_ * 64 * 64 * C_ * 2);
  _Float16* q   = (_Float16*)alloc((size_t)B_ * HW_ * C_ * 2);
  float*    hd1 = (float*)alloc((size_t)B_ * HW_ * 12 * 4);
  float*    hd2 = (float*)alloc((size_t)B_ * HW_ * 12 * 4);
  _Float16* y1  = (_Float16*)alloc((size_t)B_ * HW_ * C_ * 2);
  _Float16* y2  = (_Float16*)alloc((size_t)B_ * HW_ * C_ * 2);
  float*    means = (float*)alloc((size_t)B_ * 96 * 4);
  float*    alpha = (float*)alloc((size_t)B_ * 3 * 4);

  // pools
  pool2_kernel<<<(B_ * C_ * 128 * 128 + 255) / 256, 256, 0, stream>>>(x, s1, H_, W_);
  pool2_kernel<<<(B_ * C_ * 64 * 64 + 255) / 256, 256, 0, stream>>>(s1, s2, 128, 128);
  // depthwise-separable blocks (NCHW fp32 in -> NHWC f16 out)
  ds_kernel<128><<<dim3(2, 256, B_), 256, 0, stream>>>(x, ds_dw[0], ds_pw[0], ds_g[0],
                                                       ds_b[0], ds_m[0], ds_v[0], z0, H_, W_);
  ds_kernel<128><<<dim3(1, 128, B_), 256, 0, stream>>>(s1, ds_dw[1], ds_pw[1], ds_g[1],
                                                       ds_b[1], ds_m[1], ds_v[1], z1, 128, 128);
  ds_kernel<64><<<dim3(1, 64, B_), 128, 0, stream>>>(s2, ds_dw[2], ds_pw[2], ds_g[2],
                                                     ds_b[2], ds_m[2], ds_v[2], z2, 64, 64);
  // zero mean accumulators (atomics below)
  zero_kernel<<<(B_ * 96 + 255) / 256, 256, 0, stream>>>(means, B_ * 96);
  // q projection
  pw_nhwc_kernel<<<dim3(HW_ / 128, 1, B_), 256, 0, stream>>>(z0, qproj_w, q, HW_);
  // heads
  head_kernel<<<dim3(2, 256, B_), 256, 0, stream>>>(q, h1_c1, h1_c2w, h1_c2b, hd1);
  head_kernel<<<dim3(2, 256, B_), 256, 0, stream>>>(q, h2_c1, h2_c2w, h2_c2b, hd2);
  // means & deformable sampling (y means fused)
  mean_nhwc_kernel<<<dim3(HW_ / 4096, 1, B_), 256, 0, stream>>>(z0, means, 0);
  sample_kernel<<<dim3(HW_ / 256, 1, B_), 256, 0, stream>>>(z1, 128, 128, hd1, y1, means, 32);
  sample_kernel<<<dim3(HW_ / 256, 1, B_), 256, 0, stream>>>(z2, 64, 64, hd2, y2, means, 64);
  // gate
  alpha_kernel<<<1, 32, 0, stream>>>(means, r_w1, r_b1, r_w2, r_b2, alpha);
  // final mix + pointwise + residual
  final_kernel<<<dim3(HW_ / 128, 1, B_), 256, 0, stream>>>(z0, y1, y2, alpha, final_w,
                                                           x, (float*)d_out);
}
